// Attention_12249246728638
// MI455X (gfx1250) — matrix operand order repaired and checked by reading, not compiled, not run on hardware
//
#include <hip/hip_runtime.h>
#include <hip/hip_bf16.h>

// MI455X / gfx1250, wave32. Matrix math via v_wmma_f32_16x16x32_f16,
// K/Q tile staging via global_load_async_to_lds_b128 (ASYNCcnt path).

typedef _Float16 f16;
typedef __attribute__((ext_vector_type(16))) _Float16 v16h;
typedef __attribute__((ext_vector_type(8)))  _Float16 v8h;
typedef __attribute__((ext_vector_type(4)))  _Float16 v4h;
typedef __attribute__((ext_vector_type(8)))  float    v8f;

#define B_    2
#define S_    2048
#define HID_  1024
#define H_    16
#define KVH_  4
#define D_    64
#define SCALE_ 0.125f
#define CAP_  50.0f

// ---- WMMA fragment helpers (layouts per CDNA5 ISA 7.12.2, wave32) ----
__device__ __forceinline__ v16h frag_a(const f16* p, int row0, int stride,
                                       int koff, int lane) {
  const f16* q = p + (row0 + (lane & 15)) * stride + koff + ((lane >> 4) << 3);
  union { v16h v; v8h h[2]; } u;
  u.h[0] = *(const v8h*)(q);
  u.h[1] = *(const v8h*)(q + 16);
  return u.v;
}
__device__ __forceinline__ v16h frag_b(const f16* p, int col0, int stride,
                                       int koff, int lane) {
  const f16* q = p + (col0 + (lane & 15)) * stride + koff + ((lane >> 4) << 3);
  union { v16h v; v8h h[2]; } u;
  u.h[0] = *(const v8h*)(q);
  u.h[1] = *(const v8h*)(q + 16);
  return u.v;
}
__device__ __forceinline__ v8f wmma_f16(v16h a, v16h b, v8f c) {
  return __builtin_amdgcn_wmma_f32_16x16x32_f16(false, a, false, b,
                                                (short)0, c, false, false);
}

// Async global->LDS copy of 16 bytes per lane (GVS mode: SGPR base + per-lane
// 32-bit byte offset; LDS dest = low 32 bits of flat shared pointer).
__device__ __forceinline__ void async_b128(void* lds_ptr, const void* gbase,
                                           unsigned gbyte_off) {
  unsigned loff = (unsigned)(size_t)lds_ptr;
  asm volatile("global_load_async_to_lds_b128 %0, %1, %2"
               :: "v"(loff), "v"(gbyte_off), "s"(gbase)
               : "memory");
}
__device__ __forceinline__ void wait_async0() {
  asm volatile("s_wait_asynccnt 0x0" ::: "memory");
}

// ---- GEMM: Out[M,N] = A[M,K] @ W[N,K]^T + bias, fp32 in/out, f16 WMMA ----
// Tile 128x64, BK=32, 256 threads = 8 waves in 4(m) x 2(n) grid.
__global__ __launch_bounds__(256) void gemm_nt_bias(
    const float* __restrict__ A, const float* __restrict__ W,
    const float* __restrict__ bias, float* __restrict__ Out,
    int M, int N, int K) {
  __shared__ f16 sA[128][40];
  __shared__ f16 sB[64][40];
  const int tid  = threadIdx.x;
  const int lane = tid & 31;
  const int wv   = tid >> 5;
  const int wm   = wv & 3;
  const int wn   = wv >> 2;
  const int rowBase = blockIdx.y * 128;
  const int colBase = blockIdx.x * 64;

  v8f acc[2][2] = {};

  for (int k0 = 0; k0 < K; k0 += 32) {
    if (k0 + 32 < K) {   // prefetch next K-tile into GL2 (global_prefetch_b8)
      __builtin_prefetch(A + (unsigned)((rowBase + (tid >> 1)) * K + k0 + 32) +
                         ((tid & 1) << 4), 0, 1);
      __builtin_prefetch(W + (unsigned)((colBase + (tid & 63)) * K + k0 + 32),
                         0, 1);
    }
    __syncthreads();
#pragma unroll
    for (int i = 0; i < 4; ++i) {          // A tile: 128x32 fp32 -> f16 LDS
      int slot = tid + (i << 8);
      int r = slot >> 3, kc = (slot & 7) << 2;
      float4 f = *(const float4*)(A + (unsigned)((rowBase + r) * K + k0 + kc));
      v4h h; h[0]=(f16)f.x; h[1]=(f16)f.y; h[2]=(f16)f.z; h[3]=(f16)f.w;
      *(v4h*)&sA[r][kc] = h;
    }
#pragma unroll
    for (int i = 0; i < 2; ++i) {          // B tile: 64x32 (rows of W)
      int slot = tid + (i << 8);
      int r = slot >> 3, kc = (slot & 7) << 2;
      float4 f = *(const float4*)(W + (unsigned)((colBase + r) * K + k0 + kc));
      v4h h; h[0]=(f16)f.x; h[1]=(f16)f.y; h[2]=(f16)f.z; h[3]=(f16)f.w;
      *(v4h*)&sB[r][kc] = h;
    }
    __syncthreads();
    v16h a0 = frag_a(&sA[0][0], wm * 32,      40, 0, lane);
    v16h a1 = frag_a(&sA[0][0], wm * 32 + 16, 40, 0, lane);
    v16h b0 = frag_b(&sB[0][0], wn * 32,      40, 0, lane);
    v16h b1 = frag_b(&sB[0][0], wn * 32 + 16, 40, 0, lane);
    acc[0][0] = wmma_f16(a0, b0, acc[0][0]);
    acc[0][1] = wmma_f16(a0, b1, acc[0][1]);
    acc[1][0] = wmma_f16(a1, b0, acc[1][0]);
    acc[1][1] = wmma_f16(a1, b1, acc[1][1]);
  }
  // D layout: lane L -> n=L%16, rows r + 8*(L/16)
#pragma unroll
  for (int mi = 0; mi < 2; ++mi)
#pragma unroll
    for (int ni = 0; ni < 2; ++ni) {
      int n  = colBase + wn * 32 + ni * 16 + (lane & 15);
      float bn = bias[n];
      int rb = rowBase + wm * 32 + mi * 16 + ((lane >> 4) << 3);
#pragma unroll
      for (int r = 0; r < 8; ++r)
        Out[(unsigned)((rb + r) * N + n)] = acc[mi][ni][r] + bn;
    }
}

// ---- RoPE + pack fp32 [B*S, heads*64] -> f16 [B, heads, S, 64] ----
__global__ __launch_bounds__(256) void rope_cast(
    const float* __restrict__ In, const float* __restrict__ cosb,
    const float* __restrict__ sinb, f16* __restrict__ Outh,
    int heads, int do_rope) {
  unsigned idx = blockIdx.x * 256 + threadIdx.x; // ((b*heads+hh)*S + s)*64 + d
  int d  = idx & 63;
  unsigned t  = idx >> 6;
  int s  = t & (S_ - 1);
  unsigned t2 = t >> 11;
  int hh = t2 % heads;
  int b  = t2 / heads;
  int ncols = heads * 64;
  const float* rowp = In + (unsigned)((b * S_ + s) * ncols + hh * 64);
  float v   = rowp[d];
  float out = v;
  if (do_rope) {
    float pv = (d < 32) ? -rowp[d + 32] : rowp[d - 32];
    unsigned ci = ((unsigned)(b * S_ + s) << 6) + d;
    out = v * cosb[ci] + pv * sinb[ci];
  }
  Outh[idx] = (f16)out;
}

// ---- Fused attention: per (b,h,q-tile of 64) two-pass softcap softmax ----
__global__ __launch_bounds__(256) void attn_fused(
    const f16* __restrict__ Qh, const f16* __restrict__ Kh,
    const f16* __restrict__ Vh, float* __restrict__ attn,
    float* __restrict__ ctxout) {
  __shared__ f16 sQ[64][72];
  __shared__ f16 sK[64][72];
  __shared__ f16 sVt[64][72];   // V transposed: [d][k]
  __shared__ f16 sP[64][72];
  __shared__ float sSum[64];

  const int tid  = threadIdx.x;
  const int lane = tid & 31;
  const int wv   = tid >> 5;
  const int qw   = wv & 3;    // 16-row q sub-tile
  const int kw   = wv >> 2;   // 32-col k half (scores) / 32-col d half (ctx)
  const int qt   = blockIdx.x;
  const int h    = blockIdx.y;
  const int b    = blockIdx.z;
  const int hk   = h >> 2;    // h / N_REP
  const int halfr = (lane >> 4) << 3;

  const int trow = tid >> 2, tpart = (tid & 3) << 4;        // 16-B chunk map
  const unsigned tile_goff = (unsigned)(trow * 64 + tpart) * 2;  // bytes

  {  // async load Q tile (64x64 f16) straight into LDS
    const f16* qg = Qh + (unsigned)(((b * H_ + h) * S_) + qt * 64) * 64;
    async_b128(&sQ[trow][tpart], qg, tile_goff);
    wait_async0();
  }

  float psum[8];
#pragma unroll
  for (int r = 0; r < 8; ++r) psum[r] = 0.f;

  const f16* kvbase_K = Kh + (unsigned)(((b * KVH_ + hk) * S_)) * 64;
  const f16* kvbase_V = Vh + (unsigned)(((b * KVH_ + hk) * S_)) * 64;

  // ---------- pass 1: row sums ----------
  for (int kt = 0; kt <= qt; ++kt) {
    __syncthreads();
    async_b128(&sK[trow][tpart], kvbase_K + (unsigned)(kt * 64) * 64,
               tile_goff);
    wait_async0();
    __syncthreads();
    v8f c0 = {}, c1 = {};
#pragma unroll
    for (int d0 = 0; d0 < 64; d0 += 32) {
      v16h a  = frag_a(&sQ[0][0], qw * 16, 72, d0, lane);
      v16h b0 = frag_b(&sK[0][0], kw * 32,      72, d0, lane);
      v16h b1 = frag_b(&sK[0][0], kw * 32 + 16, 72, d0, lane);
      c0 = wmma_f16(a, b0, c0);
      c1 = wmma_f16(a, b1, c1);
    }
    int qbase = qt * 64 + qw * 16 + halfr;
#pragma unroll
    for (int kc = 0; kc < 2; ++kc) {
      int kg = kt * 64 + kw * 32 + kc * 16 + (lane & 15);
#pragma unroll
      for (int r = 0; r < 8; ++r) {
        float s = (kc ? c1[r] : c0[r]) * SCALE_;
        s = fminf(fmaxf(s, -CAP_), CAP_);
        if (kg > qbase + r) s = -1.0e9f;   // exp -> 0, matches reference
        psum[r] += __expf(s);
      }
    }
  }

  __syncthreads();
  if (tid < 64) sSum[tid] = 0.f;
  __syncthreads();
#pragma unroll
  for (int r = 0; r < 8; ++r)
    atomicAdd(&sSum[qw * 16 + halfr + r], psum[r]);
  __syncthreads();
  if (tid < 64) sSum[tid] = 1.0f / sSum[tid];
  __syncthreads();

  v8f ctx0 = {}, ctx1 = {};

  // ---------- pass 2: normalized attn + ctx ----------
  for (int kt = 0; kt <= qt; ++kt) {
    __syncthreads();
    async_b128(&sK[trow][tpart], kvbase_K + (unsigned)(kt * 64) * 64,
               tile_goff);
    {  // V tile -> VGPRs -> transposed scatter into sVt[d][k]
      const f16* vptr = kvbase_V + (unsigned)(kt * 64) * 64;
      int kl = tid >> 2, db = (tid & 3) << 4;
      union { uint4 u; f16 e[8]; } u0, u1;
      u0.u = *(const uint4*)(vptr + (unsigned)(kl * 64 + db));
      u1.u = *(const uint4*)(vptr + (unsigned)(kl * 64 + db + 8));
#pragma unroll
      for (int j = 0; j < 8; ++j) {
        sVt[db + j][kl]     = u0.e[j];
        sVt[db + 8 + j][kl] = u1.e[j];
      }
    }
    wait_async0();
    __syncthreads();
    v8f c0 = {}, c1 = {};
#pragma unroll
    for (int d0 = 0; d0 < 64; d0 += 32) {
      v16h a  = frag_a(&sQ[0][0], qw * 16, 72, d0, lane);
      v16h b0 = frag_b(&sK[0][0], kw * 32,      72, d0, lane);
      v16h b1 = frag_b(&sK[0][0], kw * 32 + 16, 72, d0, lane);
      c0 = wmma_f16(a, b0, c0);
      c1 = wmma_f16(a, b1, c1);
    }
#pragma unroll
    for (int kc = 0; kc < 2; ++kc) {
      int klocal = kw * 32 + kc * 16 + (lane & 15);
      int kg = kt * 64 + klocal;
#pragma unroll
      for (int r = 0; r < 8; ++r) {
        int qlocal = qw * 16 + halfr + r;
        int qg = qt * 64 + qlocal;
        float s = (kc ? c1[r] : c0[r]) * SCALE_;
        s = fminf(fmaxf(s, -CAP_), CAP_);
        if (kg > qg) s = -1.0e9f;
        float p = __expf(s) * sSum[qlocal];
        attn[(unsigned)(((b * H_ + h) * S_) + qg) * S_ + kg] = p;
        sP[qlocal][klocal] = (f16)p;
      }
    }
    __syncthreads();
#pragma unroll
    for (int k0 = 0; k0 < 64; k0 += 32) {  // ctx += P @ V  (wave: q x 32 d)
      v16h a  = frag_a(&sP[0][0], qw * 16, 72, k0, lane);
      v16h b0 = frag_b(&sVt[0][0], kw * 32,      72, k0, lane);
      v16h b1 = frag_b(&sVt[0][0], kw * 32 + 16, 72, k0, lane);
      ctx0 = wmma_f16(a, b0, ctx0);
      ctx1 = wmma_f16(a, b1, ctx1);
    }
  }

  // zero-fill strictly-causal upper tiles of attn (skipped all compute there)
  int kstart = (qt + 1) * 64;
  if (kstart < S_) {
    int nv = (S_ - kstart) >> 2;
    float4 z = make_float4(0.f, 0.f, 0.f, 0.f);
    for (int i = tid; i < 64 * nv; i += 256) {
      int row = i / nv, c4 = i % nv;
      *(float4*)(attn + (unsigned)(((b * H_ + h) * S_) + qt * 64 + row) * S_ +
                 kstart + (c4 << 2)) = z;
    }
  }

  // write ctx fp32 scratch: row = b*S+q, col = h*64+d (ready for out-proj GEMM)
#pragma unroll
  for (int ci = 0; ci < 2; ++ci) {
    int dcol = kw * 32 + ci * 16 + (lane & 15);
    int rb   = qt * 64 + qw * 16 + halfr;
#pragma unroll
    for (int r = 0; r < 8; ++r) {
      float v = ci ? ctx1[r] : ctx0[r];
      ctxout[(unsigned)((b * S_ + rb + r) * 1024 + h * 64 + dcol)] = v;
    }
  }
}

extern "C" void kernel_launch(void* const* d_in, const int* in_sizes, int n_in,
                              void* d_out, int out_size, void* d_ws, size_t ws_size,
                              hipStream_t stream) {
  (void)in_sizes; (void)n_in; (void)out_size; (void)ws_size;
  const float* X    = (const float*)d_in[0];
  const float* cosb = (const float*)d_in[1];
  const float* sinb = (const float*)d_in[2];
  // d_in[3] = mask (causal tril) -- implicit in kernel
  const float* Wq = (const float*)d_in[4];
  const float* bq = (const float*)d_in[5];
  const float* Wk = (const float*)d_in[6];
  const float* bk = (const float*)d_in[7];
  const float* Wv = (const float*)d_in[8];
  const float* bv = (const float*)d_in[9];
  const float* Wo = (const float*)d_in[10];
  const float* bo = (const float*)d_in[11];

  // workspace layout (~52 MB total)
  char*  ws  = (char*)d_ws;
  float* Qf  = (float*)(ws);                       // 4096x1024 f32 (16 MB)
  float* Kf  = (float*)(ws + (size_t)(16u << 20)); // 4096x256  (4 MB)
  float* Vf  = (float*)(ws + (size_t)(20u << 20)); // 4096x256  (4 MB)
  float* Ctx = (float*)(ws + (size_t)(24u << 20)); // 4096x1024 (16 MB)
  f16*   Qh  = (f16*)  (ws + (size_t)(40u << 20)); // [B,H,S,64]  (8 MB)
  f16*   Khp = (f16*)  (ws + (size_t)(48u << 20)); // [B,KVH,S,64] (2 MB)
  f16*   Vhp = (f16*)  (ws + (size_t)(50u << 20)); // [B,KVH,S,64] (2 MB)

  float* outp  = (float*)d_out;
  float* attnp = outp + (size_t)B_ * S_ * HID_;

  gemm_nt_bias<<<dim3(1024 / 64, 4096 / 128), 256, 0, stream>>>(
      X, Wq, bq, Qf, 4096, 1024, 1024);
  gemm_nt_bias<<<dim3(256 / 64, 4096 / 128), 256, 0, stream>>>(
      X, Wk, bk, Kf, 4096, 256, 1024);
  gemm_nt_bias<<<dim3(256 / 64, 4096 / 128), 256, 0, stream>>>(
      X, Wv, bv, Vf, 4096, 256, 1024);

  rope_cast<<<(B_ * H_ * S_ * 64) / 256, 256, 0, stream>>>(
      Qf, cosb, sinb, Qh, H_, 1);
  rope_cast<<<(B_ * KVH_ * S_ * 64) / 256, 256, 0, stream>>>(
      Kf, cosb, sinb, Khp, KVH_, 1);
  rope_cast<<<(B_ * KVH_ * S_ * 64) / 256, 256, 0, stream>>>(
      Vf, cosb, sinb, Vhp, KVH_, 0);

  attn_fused<<<dim3(S_ / 64, H_, B_), 256, 0, stream>>>(
      Qh, Khp, Vhp, attnp, Ctx);

  gemm_nt_bias<<<dim3(1024 / 64, 4096 / 128), 256, 0, stream>>>(
      Ctx, Wo, bo, outp, 4096, 1024, 1024);
}
